// USA_90134183674296
// MI455X (gfx1250) — matrix operand order repaired and checked by reading, not compiled, not run on hardware
//
#include <hip/hip_runtime.h>
#include <hip/hip_bf16.h>

typedef __attribute__((ext_vector_type(16))) _Float16 v16h;
typedef __attribute__((ext_vector_type(8)))  _Float16 v8h;
typedef __attribute__((ext_vector_type(4)))  _Float16 v4h;
typedef __attribute__((ext_vector_type(8)))  float    v8f;
typedef __attribute__((ext_vector_type(4)))  unsigned int u32x4;
typedef __attribute__((ext_vector_type(8)))  int          i32x8;
typedef __attribute__((ext_vector_type(4)))  int          i32x4;

#define S_LEN 2048
#define DIN   128
#define IDIM  128
#define FDIM  32
#define ROW   136   // padded LDS row stride in halves (272B = 68 dwords -> conflict-free b128 column reads)

// LDS layout (bytes). Core region identical with/without TDM so host shmem size is unconditional.
#define LDS_CORE_BYTES ((3 * 128 * ROW + 32 * ROW) * 2 + (128 + 128 + 32) * 4) // 114304
#define STAGE_A_OFF    LDS_CORE_BYTES                                          // 64KB fp32 stage
#define STAGE_B_OFF    (STAGE_A_OFF + 65536)                                   // 64KB fp32 stage
#define LDS_TOTAL_BYTES (STAGE_B_OFF + 65536)                                  // 245376 < 320KB/WGP

#if defined(__HIP_DEVICE_COMPILE__) && __has_builtin(__builtin_amdgcn_tensor_load_to_lds)
#define HAVE_TDM 1
#endif

__device__ __forceinline__ v16h cat16(v8h lo, v8h hi) {
  return __builtin_shufflevector(lo, hi, 0,1,2,3,4,5,6,7,8,9,10,11,12,13,14,15);
}

// Cheap activations: single v_exp_f32 + v_rcp_f32 instead of IEEE-divide expansion.
__device__ __forceinline__ float fast_sigmoid(float x) {
  return __builtin_amdgcn_rcpf(1.0f + __expf(-x));
}
__device__ __forceinline__ float fast_silu(float x) {
  return x * __builtin_amdgcn_rcpf(1.0f + __expf(-x));
}
__device__ __forceinline__ float fast_tanh(float x) {
  return 1.0f - 2.0f * __builtin_amdgcn_rcpf(1.0f + __expf(2.0f * x));
}

#if defined(HAVE_TDM)
// Tensor Data Mover: DMA a dim1 x dim0 fp32 tile (row stride stride0 elements) into LDS.
// D# per CDNA5 ISA: group0 = {count=1, lds_addr, global_addr, type=2},
// group1 = {data_size=4B, tensor_dim0/1, tile_dim0/1, tensor_dim0_stride}.
// Toolchain here is the 6-arg clang-23 form: (u32x4, i32x8, i32x4, i32x4, i32x8, i32 cpol).
__device__ __forceinline__ void tdm_load_tile_f32(const float* gsrc, unsigned lds_byte_off,
                                                  int dim0, int dim1, int stride0) {
  unsigned long long ga = (unsigned long long)(size_t)gsrc;
  u32x4 g0;
  g0[0] = 1u;                                                   // count=1, user mode, no gather
  g0[1] = __builtin_amdgcn_groupstaticsize() + lds_byte_off;    // lds_addr (bytes)
  g0[2] = (unsigned)ga;                                         // global_addr[31:0]
  g0[3] = (unsigned)((ga >> 32) & 0x01FFFFFFull) | (2u << 30);  // global_addr[56:32] | type=2
  i32x8 g1;
  g1[0] = (int)(2u << 16);                                      // data_size = 2 -> 4 bytes
  g1[1] = (int)((unsigned)(dim0 & 0xFFFF) << 16);               // tensor_dim0[15:0]
  g1[2] = (int)(((unsigned)dim0 >> 16) |
                ((unsigned)(dim1 & 0xFFFF) << 16));             // tensor_dim0[31:16] | tensor_dim1[15:0]
  g1[3] = (int)(((unsigned)dim1 >> 16) |
                ((unsigned)(dim0 & 0xFFFF) << 16));             // tensor_dim1[31:16] | tile_dim0
  g1[4] = (int)(dim1 & 0xFFFF);                                 // tile_dim1 (tile_dim2 = 0)
  g1[5] = stride0;                                              // tensor_dim0_stride[31:0]
  g1[6] = 0;                                                    // stride0[47:32] | stride1[15:0]
  g1[7] = 0;
  i32x4 gz4 = {0, 0, 0, 0};
  i32x8 gz8 = {0, 0, 0, 0, 0, 0, 0, 0};
  __builtin_amdgcn_tensor_load_to_lds(g0, g1, gz4, gz4, gz8, 0);
}
#endif

// Convert a 128x128 fp32 block (src: LDS stage or global) into transposed f16 weights.
__device__ __forceinline__ void cvt_w128(const float* __restrict__ src,
                                         _Float16* __restrict__ dst, int t) {
#pragma unroll
  for (int i = 0; i < 16; ++i) {
    const int flat = (i * 256 + t) * 4;
    float4 v = *(const float4*)(src + flat);
    const int k = flat >> 7, n = flat & 127;
    dst[(n + 0) * ROW + k] = (_Float16)v.x;
    dst[(n + 1) * ROW + k] = (_Float16)v.y;
    dst[(n + 2) * ROW + k] = (_Float16)v.z;
    dst[(n + 3) * ROW + k] = (_Float16)v.w;
  }
}

// Convert a 128x128 fp32 block into row-major f16 activations.
__device__ __forceinline__ void cvt_x128(const float* __restrict__ src,
                                         _Float16* __restrict__ dst, int t) {
#pragma unroll
  for (int i = 0; i < 16; ++i) {
    const int flat = (i * 256 + t) * 4;
    float4 v = *(const float4*)(src + flat);
    const int row = flat >> 7, c = flat & 127;
    v4h p = {(_Float16)v.x, (_Float16)v.y, (_Float16)v.z, (_Float16)v.w};
    *(v4h*)(dst + row * ROW + c) = p;
  }
}

// One 16-row strip (per wave) of a [*,128] @ [128,128] layer with bias + SiLU.
__device__ __forceinline__ void mlp_layer_128(const _Float16* sIn, const _Float16* sWt,
                                              const float* sBias, _Float16* sOut,
                                              int rbase, int lrow, int lhi) {
  const int arow = rbase + lrow;
#pragma unroll
  for (int nt = 0; nt < 8; ++nt) {
    const int n = nt * 16 + lrow;
    const float bb = sBias[n];
    v8f acc;
#pragma unroll
    for (int r = 0; r < 8; ++r) acc[r] = bb;
#pragma unroll
    for (int kt = 0; kt < 4; ++kt) {
      const int k0 = kt * 32;
      v8h alo = *(const v8h*)(sIn + arow * ROW + k0 + lhi * 16);
      v8h ahi = *(const v8h*)(sIn + arow * ROW + k0 + lhi * 16 + 8);
      v16h a  = cat16(alo, ahi);
      v8h blo = *(const v8h*)(sWt + n * ROW + k0 + lhi * 16);
      v8h bhi = *(const v8h*)(sWt + n * ROW + k0 + lhi * 16 + 8);
      v16h bf = cat16(blo, bhi);
      acc = __builtin_amdgcn_wmma_f32_16x16x32_f16(false, a, false, bf,
                                                   (short)0, acc, false, false);
    }
#pragma unroll
    for (int r = 0; r < 8; ++r) {
      sOut[(rbase + r + 8 * lhi) * ROW + nt * 16 + lrow] = (_Float16)fast_silu(acc[r]);
    }
  }
}

// Phase 1: per-head 3-layer MLP + tanh lift for K (z=0) and Q (z=1).
// grid = (16 M-tiles, 64 bh, 2), block = 256 (8 waves, each owns a 16-row strip).
__global__ __launch_bounds__(256) void lift_kernel(
    const float* __restrict__ Kin, const float* __restrict__ Qin,
    const float* __restrict__ wk1, const float* __restrict__ bk1,
    const float* __restrict__ wk2, const float* __restrict__ bk2,
    const float* __restrict__ wk3, const float* __restrict__ bk3,
    const float* __restrict__ wq1, const float* __restrict__ bq1,
    const float* __restrict__ wq2, const float* __restrict__ bq2,
    const float* __restrict__ wq3, const float* __restrict__ bq3,
    float* __restrict__ kliftF32,
    _Float16* __restrict__ klift16, _Float16* __restrict__ qlift16) {
  extern __shared__ char smem[];
  _Float16* sW  = (_Float16*)smem;            // 128 x ROW  (W1, later W2, transposed)
  _Float16* sA  = sW  + 128 * ROW;            // 128 x ROW  (X, later H2)
  _Float16* sH  = sA  + 128 * ROW;            // 128 x ROW  (H1)
  _Float16* sW3 = sH  + 128 * ROW;            // 32  x ROW  (W3 transposed)
  float*    sB1 = (float*)(sW3 + 32 * ROW);
  float*    sB2 = sB1 + 128;
  float*    sB3 = sB2 + 128;
#if defined(HAVE_TDM)
  const float* stageA = (const float*)(smem + STAGE_A_OFF);
  const float* stageB = (const float*)(smem + STAGE_B_OFF);
#endif

  const int tileM = blockIdx.x;
  const int bh    = blockIdx.y;
  const int head  = bh & 31;
  const int isQ   = blockIdx.z;

  const float* X  = isQ ? Qin : Kin;
  const float* w1 = isQ ? wq1 : wk1;  const float* b1 = isQ ? bq1 : bk1;
  const float* w2 = isQ ? wq2 : wk2;  const float* b2 = isQ ? bq2 : bk2;
  const float* w3 = isQ ? wq3 : wk3;  const float* b3 = isQ ? bq3 : bk3;
  _Float16* lift16 = isQ ? qlift16 : klift16;

  const int t0 = tileM * 128;
  const float* Xm  = X  + ((size_t)bh * S_LEN + t0) * DIN;
  const float* w1h = w1 + (size_t)head * DIN * IDIM;
  const float* w2h = w2 + (size_t)head * IDIM * IDIM;
  const float* w3h = w3 + (size_t)head * IDIM * FDIM;

  const int t = threadIdx.x;

#if defined(HAVE_TDM)
  // Kick off W1 and X-tile DMAs immediately (one issuing wave; TDM ignores EXEC).
  if (t < 32) {
    tdm_load_tile_f32(w1h, STAGE_A_OFF, 128, 128, 128);
    tdm_load_tile_f32(Xm,  STAGE_B_OFF, 128, 128, 128);
  }
#endif

  // Small operands via regular loads while DMAs fly.
  if (t < 128) { sB1[t] = b1[head * IDIM + t]; sB2[t] = b2[head * IDIM + t]; }
  if (t < 32)  { sB3[t] = b3[head * FDIM + t]; }
#pragma unroll
  for (int i = 0; i < 4; ++i) {          // W3 -> sW3 transposed
    const int flat = (i * 256 + t) * 4;
    float4 v = *(const float4*)(w3h + flat);
    const int k = flat >> 5, n = flat & 31;
    sW3[(n + 0) * ROW + k] = (_Float16)v.x;
    sW3[(n + 1) * ROW + k] = (_Float16)v.y;
    sW3[(n + 2) * ROW + k] = (_Float16)v.z;
    sW3[(n + 3) * ROW + k] = (_Float16)v.w;
  }

#if defined(HAVE_TDM)
  if (t < 32) __builtin_amdgcn_s_wait_tensorcnt(1);  // TDM in-order: W1 (1st issue) done
  __syncthreads();
  cvt_w128(stageA, sW, t);                           // W1 fp32(LDS) -> f16 transposed
  __syncthreads();                                   // stageA free
  if (t < 32) {
    tdm_load_tile_f32(w2h, STAGE_A_OFF, 128, 128, 128);  // W2 DMA overlaps layer 1
    __builtin_amdgcn_s_wait_tensorcnt(1);                // X done (W2 may be in flight)
  }
  __syncthreads();
  cvt_x128(stageB, sA, t);                           // X fp32(LDS) -> f16 row-major
#else
  cvt_w128(w1h, sW, t);
  cvt_x128(Xm, sA, t);
#endif
  __syncthreads();

  const int wave = t >> 5, lane = t & 31, lrow = lane & 15, lhi = lane >> 4;
  const int rbase = wave * 16;

  // Layer 1: H1 = silu(X @ W1 + b1)
  mlp_layer_128(sA, sW, sB1, sH, rbase, lrow, lhi);
  __syncthreads();

#if defined(HAVE_TDM)
  if (t < 32) __builtin_amdgcn_s_wait_tensorcnt(0);  // W2 arrived
  __syncthreads();
  cvt_w128(stageA, sW, t);
#else
  cvt_w128(w2h, sW, t);
#endif
  __syncthreads();

  // Layer 2: H2 = silu(H1 @ W2 + b2)  (into sA; X no longer needed)
  mlp_layer_128(sH, sW, sB2, sA, rbase, lrow, lhi);
  __syncthreads();

  // Layer 3: lift = tanh(H2 @ W3 + b3), N = 32
  {
    const int arow = rbase + lrow;
#pragma unroll
    for (int nt = 0; nt < 2; ++nt) {
      const int n = nt * 16 + lrow;
      const float bb = sB3[n];
      v8f acc;
#pragma unroll
      for (int r = 0; r < 8; ++r) acc[r] = bb;
#pragma unroll
      for (int kt = 0; kt < 4; ++kt) {
        const int k0 = kt * 32;
        v8h alo = *(const v8h*)(sA + arow * ROW + k0 + lhi * 16);
        v8h ahi = *(const v8h*)(sA + arow * ROW + k0 + lhi * 16 + 8);
        v16h a  = cat16(alo, ahi);
        v8h blo = *(const v8h*)(sW3 + n * ROW + k0 + lhi * 16);
        v8h bhi = *(const v8h*)(sW3 + n * ROW + k0 + lhi * 16 + 8);
        v16h bf = cat16(blo, bhi);
        acc = __builtin_amdgcn_wmma_f32_16x16x32_f16(false, a, false, bf,
                                                     (short)0, acc, false, false);
      }
#pragma unroll
      for (int r = 0; r < 8; ++r) {
        const float val = fast_tanh(acc[r]);
        const int row = t0 + rbase + r + 8 * lhi;
        const size_t idx = ((size_t)bh * S_LEN + row) * FDIM + n;
        lift16[idx] = (_Float16)val;
        if (!isQ) kliftF32[idx] = val;
      }
    }
  }
}

// Phase 2: scores = sigmoid(causal(Qlift @ Klift^T)).  K-dim = 32 -> 1 WMMA/tile.
// grid = (8 col-blocks of 256, 16 row-blocks of 128, 64 bh), block = 256 (8 waves).
__global__ __launch_bounds__(256) void scores_kernel(
    const _Float16* __restrict__ q16, const _Float16* __restrict__ k16,
    float* __restrict__ out) {
  const int bh   = blockIdx.z;
  const int t    = threadIdx.x;
  const int wave = t >> 5, lane = t & 31, lrow = lane & 15, lhi = lane >> 4;
  const int trow0 = blockIdx.y * 128 + wave * 16;
  const int scol0 = blockIdx.x * 256;

  const _Float16* qb = q16 + (size_t)bh * S_LEN * FDIM;
  const _Float16* kb = k16 + (size_t)bh * S_LEN * FDIM;
  float* ob = out + (size_t)bh * S_LEN * S_LEN;

  // A fragment: 16 Qlift rows of this wave's strip (constant across col tiles)
  const int arow = trow0 + lrow;
  v8h alo = *(const v8h*)(qb + (size_t)arow * FDIM + lhi * 16);
  v8h ahi = *(const v8h*)(qb + (size_t)arow * FDIM + lhi * 16 + 8);
  v16h a  = cat16(alo, ahi);

  for (int nt = 0; nt < 16; ++nt) {
    const int sc = scol0 + nt * 16;

    if (sc > trow0 + 15) {                 // fully above the diagonal: sigmoid(f32_min)=0
#pragma unroll
      for (int r = 0; r < 8; ++r)
        ob[(size_t)(trow0 + r + 8 * lhi) * S_LEN + sc + lrow] = 0.0f;
      continue;                            // wave-uniform branch: EXEC stays all-ones
    }

    if (sc + 256 < S_LEN)
      __builtin_prefetch(kb + (size_t)(sc + 256) * FDIM, 0, 0);

    // B fragment: 16 Klift rows (row-major f16 rows ARE the WMMA B layout)
    const int n = sc + lrow;
    v8h blo = *(const v8h*)(kb + (size_t)n * FDIM + lhi * 16);
    v8h bhi = *(const v8h*)(kb + (size_t)n * FDIM + lhi * 16 + 8);
    v16h bf = cat16(blo, bhi);

    v8f acc = {0.f, 0.f, 0.f, 0.f, 0.f, 0.f, 0.f, 0.f};
    acc = __builtin_amdgcn_wmma_f32_16x16x32_f16(false, a, false, bf,
                                                 (short)0, acc, false, false);

    const bool full = (sc + 15 <= trow0);  // entire tile on/below diagonal
#pragma unroll
    for (int r = 0; r < 8; ++r) {
      const int row = trow0 + r + 8 * lhi;
      float v = fast_sigmoid(acc[r]);
      if (!full && (sc + lrow > row)) v = 0.0f;
      ob[(size_t)row * S_LEN + sc + lrow] = v;
    }
  }
}

extern "C" void kernel_launch(void* const* d_in, const int* in_sizes, int n_in,
                              void* d_out, int out_size, void* d_ws, size_t ws_size,
                              hipStream_t stream) {
  (void)in_sizes; (void)n_in; (void)out_size; (void)ws_size;
  const float* K   = (const float*)d_in[0];
  const float* Q   = (const float*)d_in[1];
  const float* wk1 = (const float*)d_in[2];  const float* bk1 = (const float*)d_in[3];
  const float* wk2 = (const float*)d_in[4];  const float* bk2 = (const float*)d_in[5];
  const float* wk3 = (const float*)d_in[6];  const float* bk3 = (const float*)d_in[7];
  const float* wq1 = (const float*)d_in[8];  const float* bq1 = (const float*)d_in[9];
  const float* wq2 = (const float*)d_in[10]; const float* bq2 = (const float*)d_in[11];
  const float* wq3 = (const float*)d_in[12]; const float* bq3 = (const float*)d_in[13];

  const size_t BH = 64;
  float* scores    = (float*)d_out;
  float* kliftF32  = scores + BH * (size_t)S_LEN * S_LEN;     // Klift fp32 output
  _Float16* k16    = (_Float16*)d_ws;                          // Klift f16 (8 MB)
  _Float16* q16    = k16 + BH * (size_t)S_LEN * FDIM;          // Qlift f16 (8 MB)

  const size_t shmem = LDS_TOTAL_BYTES;                        // stages always reserved
  (void)hipFuncSetAttribute((const void*)lift_kernel,
                            hipFuncAttributeMaxDynamicSharedMemorySize, (int)shmem);

  lift_kernel<<<dim3(16, 64, 2), 256, shmem, stream>>>(
      K, Q, wk1, bk1, wk2, bk2, wk3, bk3, wq1, bq1, wq2, bq2, wq3, bq3,
      kliftF32, k16, q16);

  scores_kernel<<<dim3(8, 16, 64), 256, 0, stream>>>(q16, k16, scores);
}
